// DifferentialAttention_8993661518705
// MI455X (gfx1250) — compile-verified, matrix operand order checked by reading, not run on hardware
//
#include <hip/hip_runtime.h>

#define DEV __device__ __forceinline__

typedef _Float16 v16h __attribute__((ext_vector_type(16)));
typedef float    v8f  __attribute__((ext_vector_type(8)));

union Frag16 { v16h v; unsigned u[8]; _Float16 h[16]; };

DEV unsigned pack2(float a, float b) {
  union { _Float16 h[2]; unsigned u; } t;
  t.h[0] = (_Float16)a; t.h[1] = (_Float16)b;
  return t.u;
}

#define WMMA16(A, B, C) \
  __builtin_amdgcn_wmma_f32_16x16x32_f16(false, (A), false, (B), (short)0, (C), false, false)

// ---------------------------------------------------------------------------
// Generic GEMM: C[M,N] = A[M,K] * B[N,K]^T   (B always f32 weights)
// block = 256 threads = 8 waves (2x4), block tile 128x256, BK=32,
// double-buffered LDS; per wave: 64x64 output = 4x4 WMMA accumulators.
// Pattern per k-step: issue next tile's global loads -> 16 WMMAs on current
// LDS tile (hides HBM latency) -> convert+ds_store into the other buffer.
// ---------------------------------------------------------------------------
template <bool A_F16, bool OUT_F16>
__global__ __launch_bounds__(256) void gemm_wmma(
    const void* __restrict__ Aptr, const float* __restrict__ B,
    void* __restrict__ Cptr, int M, int N, int K)
{
  __shared__ _Float16 As[2][128 * 32];   // 2 x 8 KB
  __shared__ _Float16 Bs[2][256 * 32];   // 2 x 16 KB

  const int tid  = threadIdx.x;
  const int lane = tid & 31;
  const int wave = tid >> 5;
  const int wm   = wave >> 2;   // 0..1 : M direction
  const int wn   = wave & 3;    // 0..3 : N direction
  const int g    = lane >> 4;   // half-wave
  const int ln   = lane & 15;
  const int bm   = blockIdx.y * 128;
  const int bn   = blockIdx.x * 256;

  const int ra = tid >> 1;          // A staging: row 0..127
  const int ca = (tid & 1) * 16;    // A staging: col 0 or 16
  const int rb = tid;               // B staging: row 0..255 (32 elems each)

  v8f acc[4][4];
  #pragma unroll
  for (int i = 0; i < 4; i++) {
    #pragma unroll
    for (int j = 0; j < 4; j++) { v8f z = {}; acc[i][j] = z; }
  }

  // staging registers (held across the WMMA block for latency hiding)
  float4 arf[4]; uint4 arh[2]; float4 brf[8];

  auto loadA = [&](int k0) {
    if constexpr (A_F16) {
      const _Float16* Af = (const _Float16*)Aptr;
      const uint4* src = (const uint4*)(Af + (size_t)(bm + ra) * K + k0 + ca);
      arh[0] = src[0]; arh[1] = src[1];
    } else {
      const float* Af = (const float*)Aptr;
      const float4* src = (const float4*)(Af + (size_t)(bm + ra) * K + k0 + ca);
      arf[0] = src[0]; arf[1] = src[1]; arf[2] = src[2]; arf[3] = src[3];
    }
  };
  auto loadB = [&](int k0) {
    const float4* src = (const float4*)(B + (size_t)(bn + rb) * K + k0);
    #pragma unroll
    for (int i = 0; i < 8; i++) brf[i] = src[i];
  };
  auto storeA = [&](int buf) {
    _Float16* dst = &As[buf][ra * 32 + ca];
    if constexpr (A_F16) {
      *(uint4*)dst = arh[0]; *(uint4*)(dst + 8) = arh[1];
    } else {
      uint4 p0, p1;
      p0.x = pack2(arf[0].x, arf[0].y); p0.y = pack2(arf[0].z, arf[0].w);
      p0.z = pack2(arf[1].x, arf[1].y); p0.w = pack2(arf[1].z, arf[1].w);
      p1.x = pack2(arf[2].x, arf[2].y); p1.y = pack2(arf[2].z, arf[2].w);
      p1.z = pack2(arf[3].x, arf[3].y); p1.w = pack2(arf[3].z, arf[3].w);
      *(uint4*)dst = p0; *(uint4*)(dst + 8) = p1;
    }
  };
  auto storeB = [&](int buf) {
    _Float16* dst = &Bs[buf][rb * 32];
    uint4 p0, p1, p2, p3;
    p0.x = pack2(brf[0].x, brf[0].y); p0.y = pack2(brf[0].z, brf[0].w);
    p0.z = pack2(brf[1].x, brf[1].y); p0.w = pack2(brf[1].z, brf[1].w);
    p1.x = pack2(brf[2].x, brf[2].y); p1.y = pack2(brf[2].z, brf[2].w);
    p1.z = pack2(brf[3].x, brf[3].y); p1.w = pack2(brf[3].z, brf[3].w);
    p2.x = pack2(brf[4].x, brf[4].y); p2.y = pack2(brf[4].z, brf[4].w);
    p2.z = pack2(brf[5].x, brf[5].y); p2.w = pack2(brf[5].z, brf[5].w);
    p3.x = pack2(brf[6].x, brf[6].y); p3.y = pack2(brf[6].z, brf[6].w);
    p3.z = pack2(brf[7].x, brf[7].y); p3.w = pack2(brf[7].z, brf[7].w);
    *(uint4*)dst        = p0; *(uint4*)(dst + 8)  = p1;
    *(uint4*)(dst + 16) = p2; *(uint4*)(dst + 24) = p3;
  };

  const int ntiles = K / 32;

  // prologue: stage tile 0 into buffer 0
  loadA(0); loadB(0); storeA(0); storeB(0);

  for (int t = 0; t < ntiles; t++) {
    __syncthreads();                      // tile t staged; buffer t^1 drained
    const int cur  = t & 1;
    const bool more = (t + 1 < ntiles);
    if (more) {
      loadA((t + 1) * 32);                // issue next-tile global loads now
      loadB((t + 1) * 32);
      if (t + 2 < ntiles)                 // pull tile t+2 weights toward L2
        __builtin_prefetch(B + (size_t)(bn + rb) * K + (t + 2) * 32, 0, 0);
    }

    // ---- fragments + 16 WMMAs on the current LDS tile ----
    Frag16 a[4], b[4];
    #pragma unroll
    for (int i = 0; i < 4; i++) {
      const _Float16* arow = &As[cur][(wm * 64 + i * 16 + ln) * 32];
      #pragma unroll
      for (int p = 0; p < 8; p++) {
        int kb = (p & 3) * 2 + g * 8 + (p >> 2) * 16;
        a[i].u[p] = *(const unsigned*)(arow + kb);
      }
    }
    #pragma unroll
    for (int j = 0; j < 4; j++) {
      const _Float16* brow = &Bs[cur][(wn * 64 + j * 16 + ln) * 32];
      #pragma unroll
      for (int p = 0; p < 8; p++) {
        int kb = (p & 3) * 2 + g * 8 + (p >> 2) * 16;
        b[j].u[p] = *(const unsigned*)(brow + kb);
      }
    }
    #pragma unroll
    for (int i = 0; i < 4; i++) {
      #pragma unroll
      for (int j = 0; j < 4; j++)
        acc[i][j] = WMMA16(a[i].v, b[j].v, acc[i][j]);
    }

    if (more) { storeA(cur ^ 1); storeB(cur ^ 1); }  // after WMMAs: waits on loads here
  }

  // ---- epilogue ----
  #pragma unroll
  for (int i = 0; i < 4; i++) {
    #pragma unroll
    for (int j = 0; j < 4; j++) {
      #pragma unroll
      for (int v = 0; v < 8; v++) {
        int row = bm + wm * 64 + i * 16 + v + 8 * g;
        int col = bn + wn * 64 + j * 16 + ln;
        float val = acc[i][j][v];
        if constexpr (OUT_F16)
          ((_Float16*)Cptr)[(size_t)row * N + col] = (_Float16)val;
        else
          ((float*)Cptr)[(size_t)row * N + col] = val;
      }
    }
  }
}

// ---------------------------------------------------------------------------
// Differential flash attention, fused epilogue (lambda-combine + groupnorm).
// grid = (16 heads, 16 query-slabs of 128), block = 256 threads (8 waves).
// Each wave owns a 16-query tile; KV streamed in 32-key tiles via LDS.
// ---------------------------------------------------------------------------
DEV void attn_branch(const Frag16 qf[2], const _Float16* __restrict__ ks,
                     _Float16* __restrict__ pbuf, const Frag16 vb[8],
                     v8f acc[8], float m[8], float l[8],
                     int kt, int q0, int g, int ln, float sc)
{
  // scores: 16 queries x 32 keys = two C fragments, K = 64 channels (2 WMMA each)
  v8f s[2];
  #pragma unroll
  for (int kf = 0; kf < 2; kf++) {
    v8f z = {}; s[kf] = z;
    #pragma unroll
    for (int kc = 0; kc < 2; kc++) {
      Frag16 b;
      const _Float16* krow = ks + (kf * 16 + ln) * 64 + kc * 32;
      #pragma unroll
      for (int p = 0; p < 8; p++) {
        int kb = (p & 3) * 2 + g * 8 + (p >> 2) * 16;
        b.u[p] = *(const unsigned*)(krow + kb);
      }
      s[kf] = WMMA16(qf[kc].v, b.v, s[kf]);
    }
  }
  // scale (log2-domain) + causal mask + per-row max
  float rm[8];
  #pragma unroll
  for (int v = 0; v < 8; v++) {
    const int qi = q0 + v + 8 * g;
    float a0 = s[0][v] * sc;
    float a1 = s[1][v] * sc;
    if (kt + ln      > qi) a0 = -1e30f;
    if (kt + 16 + ln > qi) a1 = -1e30f;
    s[0][v] = a0; s[1][v] = a1;
    rm[v] = fmaxf(a0, a1);
  }
  #pragma unroll
  for (int off = 8; off >= 1; off >>= 1) {
    #pragma unroll
    for (int v = 0; v < 8; v++) rm[v] = fmaxf(rm[v], __shfl_xor(rm[v], off));
  }
  // online-softmax rescale of running state
  #pragma unroll
  for (int v = 0; v < 8; v++) {
    float mn   = fmaxf(m[v], rm[v]);
    float corr = exp2f(m[v] - mn);
    m[v] = mn;
    l[v] *= corr;
    #pragma unroll
    for (int f = 0; f < 8; f++) acc[f][v] *= corr;
  }
  // probabilities, row-sum, stage P (C-layout -> LDS row-major)
  float rs[8];
  #pragma unroll
  for (int v = 0; v < 8; v++) {
    float p0 = exp2f(s[0][v] - m[v]);
    float p1 = exp2f(s[1][v] - m[v]);
    rs[v] = p0 + p1;
    pbuf[(v + 8 * g) * 32 + ln]      = (_Float16)p0;
    pbuf[(v + 8 * g) * 32 + 16 + ln] = (_Float16)p1;
  }
  #pragma unroll
  for (int off = 8; off >= 1; off >>= 1) {
    #pragma unroll
    for (int v = 0; v < 8; v++) rs[v] += __shfl_xor(rs[v], off);
  }
  #pragma unroll
  for (int v = 0; v < 8; v++) l[v] += rs[v];
  // reload P in A-fragment layout (same-wave LDS ops are in-order)
  Frag16 pf;
  #pragma unroll
  for (int p = 0; p < 8; p++) {
    int kb = (p & 3) * 2 + g * 8 + (p >> 2) * 16;
    pf.u[p] = *(const unsigned*)(pbuf + ln * 32 + kb);
  }
  // acc += P * V   (V fragments shared between the two branches)
  #pragma unroll
  for (int f = 0; f < 8; f++) acc[f] = WMMA16(pf.v, vb[f].v, acc[f]);
}

__global__ __launch_bounds__(256) void diff_attn_wmma(
    const _Float16* __restrict__ qkv,   // [2048][6144] f16: Q|K|V each [h][128]
    const float* __restrict__ lq1, const float* __restrict__ lk1,
    const float* __restrict__ lq2, const float* __restrict__ lk2,
    const float* __restrict__ gamma, const float* __restrict__ beta,
    _Float16* __restrict__ attn_out)    // [2048][2048] f16
{
  __shared__ _Float16 k1s[32 * 64];      // K1 tile, key-major
  __shared__ _Float16 k2s[32 * 64];      // K2 tile, key-major
  __shared__ _Float16 vts[128 * 32];     // V tile, transposed [ch][key]
  __shared__ _Float16 pscr[8][16 * 32];  // per-wave P staging

  const int h    = blockIdx.x;
  const int qblk = blockIdx.y;
  const int tid  = threadIdx.x;
  const int lane = tid & 31;
  const int wave = tid >> 5;
  const int g    = lane >> 4;
  const int ln   = lane & 15;
  const int q0   = qblk * 128 + wave * 16;

  // Q1/Q2 A-fragments for this wave's 16 queries (2 K-chunks of 32 channels)
  Frag16 q1f[2], q2f[2];
  {
    const _Float16* qrow = qkv + (size_t)(q0 + ln) * 6144 + h * 128;
    #pragma unroll
    for (int kc = 0; kc < 2; kc++) {
      #pragma unroll
      for (int p = 0; p < 8; p++) {
        int ch = (p & 3) * 2 + g * 8 + (p >> 2) * 16 + kc * 32;
        q1f[kc].u[p] = *(const unsigned*)(qrow + ch);
        q2f[kc].u[p] = *(const unsigned*)(qrow + 64 + ch);
      }
    }
  }

  v8f acc1[8], acc2[8];
  float m1[8], l1[8], m2[8], l2[8];
  #pragma unroll
  for (int f = 0; f < 8; f++) { v8f z = {}; acc1[f] = z; acc2[f] = z; }
  #pragma unroll
  for (int v = 0; v < 8; v++) { m1[v] = -1e30f; l1[v] = 0.f; m2[v] = -1e30f; l2[v] = 0.f; }

  const float sc   = 0.125f * 1.44269504f;  // 1/sqrt(64) * log2(e)
  const int  qmax  = q0 + 15;
  const int  kend  = qblk * 128 + 128;      // causal bound for this block

  for (int kt = 0; kt < kend; kt += 32) {
    // ---- cooperative staging of K1 | K2 | V^T for 32 keys ----
    {
      const int key = tid >> 3;             // 0..31
      const int cb  = (tid & 7) * 16;       // 0..112
      const _Float16* kp = qkv + (size_t)(kt + key) * 6144 + 2048 + h * 128 + cb;
      uint4 d0 = *(const uint4*)kp;
      uint4 d1 = *(const uint4*)(kp + 8);
      if (cb < 64) {
        *(uint4*)(k1s + key * 64 + cb)     = d0;
        *(uint4*)(k1s + key * 64 + cb + 8) = d1;
      } else {
        *(uint4*)(k2s + key * 64 + (cb - 64))     = d0;
        *(uint4*)(k2s + key * 64 + (cb - 64) + 8) = d1;
      }
      const _Float16* vp = qkv + (size_t)(kt + key) * 6144 + 4096 + h * 128 + cb;
      Frag16 vt;
      *(uint4*)&vt.h[0] = *(const uint4*)vp;
      *(uint4*)&vt.h[8] = *(const uint4*)(vp + 8);
      #pragma unroll
      for (int e = 0; e < 16; e++) vts[(cb + e) * 32 + key] = vt.h[e];
    }
    __syncthreads();

    if (kt <= qmax) {   // wave-uniform causal skip
      Frag16 vb[8];     // V B-fragments (N = channel block), shared by branches
      #pragma unroll
      for (int f = 0; f < 8; f++) {
        const _Float16* vrow = vts + (f * 16 + ln) * 32;
        #pragma unroll
        for (int p = 0; p < 8; p++) {
          int kb = (p & 3) * 2 + g * 8 + (p >> 2) * 16;
          vb[f].u[p] = *(const unsigned*)(vrow + kb);
        }
      }
      attn_branch(q1f, k1s, &pscr[wave][0], vb, acc1, m1, l1, kt, q0, g, ln, sc);
      attn_branch(q2f, k2s, &pscr[wave][0], vb, acc2, m2, l2, kt, q0, g, ln, sc);
    }
    __syncthreads();
  }

  // ---- fused epilogue: A1 - lam*A2, groupnorm(128), gamma/beta, scale ----
  const float lambda_init = 0.8f - 0.6f * __expf(-3.6f);
  const float outsc = 1.0f - lambda_init;
  float lam[8], gam[8], bet[8];
  #pragma unroll
  for (int f = 0; f < 8; f++) {
    int j = f * 16 + ln;
    lam[f] = __expf(lq1[j] * lk1[j]) - __expf(lq2[j] * lk2[j]) + lambda_init;
    gam[f] = gamma[h * 128 + j];
    bet[f] = beta[h * 128 + j];
  }
  #pragma unroll
  for (int v = 0; v < 8; v++) {
    float inv1 = 1.0f / l1[v];
    float inv2 = 1.0f / l2[v];
    float o[8];
    float s1 = 0.f, s2 = 0.f;
    #pragma unroll
    for (int f = 0; f < 8; f++) {
      float t = acc1[f][v] * inv1 - lam[f] * (acc2[f][v] * inv2);
      o[f] = t; s1 += t; s2 += t * t;
    }
    #pragma unroll
    for (int off = 8; off >= 1; off >>= 1) {
      s1 += __shfl_xor(s1, off);
      s2 += __shfl_xor(s2, off);
    }
    float mu  = s1 * (1.0f / 128.0f);
    float var = s2 * (1.0f / 128.0f) - mu * mu;
    float inv = rsqrtf(var + 1e-5f);
    int row = q0 + v + 8 * g;
    _Float16* orow = attn_out + (size_t)row * 2048 + h * 128;
    #pragma unroll
    for (int f = 0; f < 8; f++) {
      float t = ((o[f] - mu) * inv * gam[f] + bet[f]) * outsc;
      orow[f * 16 + ln] = (_Float16)t;
    }
  }
}

// ---------------------------------------------------------------------------
extern "C" void kernel_launch(void* const* d_in, const int* in_sizes, int n_in,
                              void* d_out, int out_size, void* d_ws, size_t ws_size,
                              hipStream_t stream) {
  (void)in_sizes; (void)n_in; (void)out_size; (void)ws_size;
  const float* x     = (const float*)d_in[0];
  const float* Wqkv  = (const float*)d_in[1];
  const float* Wo    = (const float*)d_in[2];
  const float* lq1   = (const float*)d_in[3];
  const float* lk1   = (const float*)d_in[4];
  const float* lq2   = (const float*)d_in[5];
  const float* lk2   = (const float*)d_in[6];
  const float* gamma = (const float*)d_in[7];
  const float* beta  = (const float*)d_in[8];
  float* out = (float*)d_out;

  _Float16* qkvh  = (_Float16*)d_ws;                   // 2048*6144 f16 = 24 MB
  _Float16* attnh = qkvh + (size_t)2048 * 6144;        // 2048*2048 f16 =  8 MB

  // 1) QKV projection: (2048x2048) @ (6144x2048)^T -> f16 workspace
  gemm_wmma<false, true><<<dim3(24, 16), 256, 0, stream>>>(
      x, Wqkv, qkvh, 2048, 6144, 2048);

  // 2) fused differential flash attention + groupnorm epilogue
  diff_attn_wmma<<<dim3(16, 16), 256, 0, stream>>>(
      qkvh, lq1, lk1, lq2, lk2, gamma, beta, attnh);

  // 3) output projection: (2048x2048 f16) @ (2048x2048)^T -> f32 d_out
  gemm_wmma<true, false><<<dim3(8, 16), 256, 0, stream>>>(
      attnh, Wo, out, 2048, 2048, 2048);
}
